// Attention_26336739459073
// MI455X (gfx1250) — matrix operand order repaired and checked by reading, compiled, not run on hardware
//
#include <hip/hip_runtime.h>
#include <hip/hip_bf16.h>

// ---------------------------------------------------------------------------
// Attention block for MI455X (gfx1250, wave32, WMMA 16x16x32 bf16, fp32 acc,
// TDM tensor_load_to_lds staging of V tiles).
// B=2, S=2048, E=1024, H=16, D=64
// ---------------------------------------------------------------------------

#define BATCH   2
#define SEQ     2048
#define EMBED   1024
#define HEADS   16
#define HDIM    64
#define QKV_N   (3 * EMBED)
#define KSPLIT  4

typedef __attribute__((ext_vector_type(16))) __bf16 bf16x16;
typedef __attribute__((ext_vector_type(8)))  float  f32x8;
typedef __attribute__((ext_vector_type(4)))  float  f32x4;
typedef __attribute__((ext_vector_type(4)))  unsigned int u32x4;
typedef __attribute__((ext_vector_type(8)))  int    i32x8;
typedef __attribute__((ext_vector_type(4)))  int    i32x4;

#if __has_builtin(__builtin_amdgcn_tensor_load_to_lds)
#define USE_TDM 1
#else
#define USE_TDM 0
#endif

union BF16Pack {
    uint4   u[2];
    bf16x16 v;
};

__device__ __forceinline__ f32x8 wmma_bf16(bf16x16 a, bf16x16 b, f32x8 c) {
    // V_WMMA_F32_16X16X32_BF16 : D = A(16x32) * B(32x16) + C(16x16 f32)
    return __builtin_amdgcn_wmma_f32_16x16x32_bf16(
        /*neg_a=*/false, a, /*neg_b=*/false, b,
        /*c_mod=*/(short)0, c, /*reuse_a=*/false, /*reuse_b=*/false);
}

// A-matrix fragment (16x32, 16-bit) from fp32 row-major memory.
// Contiguous-k layout used here: lanes 0-15: M=lane, K = 0..15
//             lanes 16-31: M=lane-16, K = 16..31
__device__ __forceinline__ bf16x16 load_a_f32(const float* base, int ld, int lane) {
    int m   = lane & 15;
    int klo = (lane < 16) ? 0 : 16;
    const float* p = base + (size_t)m * ld;
    bf16x16 a;
#pragma unroll
    for (int i = 0; i < 8; ++i) {
        a[i]     = (__bf16)p[klo + i];
        a[8 + i] = (__bf16)p[klo + 8 + i];
    }
    return a;
}

// A-matrix fragment (16x32) from bf16 row-major memory, two b128 loads/lane.
__device__ __forceinline__ bf16x16 load_a_bf16(const __bf16* base, int ld, int lane) {
    int m   = lane & 15;
    int klo = (lane < 16) ? 0 : 16;
    const __bf16* p = base + (size_t)m * ld + klo;
    BF16Pack pk;
    pk.u[0] = *(const uint4*)(p);
    pk.u[1] = *(const uint4*)(p + 8);
    return pk.v;
}

// B-matrix fragment (32x16, 16-bit) from LDS fp32 tile [32][16].
// Layout: lanes 0-15: N=lane, K=0..15 ; lanes 16-31: N=lane-16, K=16..31
__device__ __forceinline__ bf16x16 load_b_lds(const float* lds, int lane) {
    int n  = lane & 15;
    int k0 = (lane < 16) ? 0 : 16;
    bf16x16 b;
#pragma unroll
    for (int i = 0; i < 16; ++i) b[i] = (__bf16)lds[(k0 + i) * 16 + n];
    return b;
}

// B fragment for scores:  B = K^T (K-dim = d, N = key index).
// Lane holds 16 contiguous d-values of one key row -> two b128 loads.
__device__ __forceinline__ bf16x16 load_b_kT(const __bf16* krows, int key_base,
                                             int dbase, int lane) {
    const __bf16* p = krows + (size_t)(key_base + (lane & 15)) * HDIM
                            + dbase + ((lane < 16) ? 0 : 16);
    BF16Pack pk;
    pk.u[0] = *(const uint4*)(p);
    pk.u[1] = *(const uint4*)(p + 8);
    return pk.v;
}

// B fragment for PV from a 32x64 bf16 tile (K-dim = key, N = d), row gather.
__device__ __forceinline__ bf16x16 load_b_v_tile(const __bf16* vtile, int dbase,
                                                 int lane) {
    int d  = dbase + (lane & 15);
    int k0 = (lane < 16) ? 0 : 16;
    bf16x16 b;
#pragma unroll
    for (int i = 0; i < 16; ++i) b[i] = vtile[(k0 + i) * HDIM + d];
    return b;
}

// ---------------------------------------------------------------------------
// Kernel 1: qkv = x @ W_qkv + b_qkv ; scatter into bf16 q/k/v workspaces
//           ([B,H,S,D]) and fp32 cached_kv output ([B,2,H,S,D]).
// ---------------------------------------------------------------------------
__global__ void __launch_bounds__(256)
qkv_gemm_kernel(const float* __restrict__ x, const float* __restrict__ Wqkv,
                const float* __restrict__ bqkv,
                unsigned short* __restrict__ qb_raw,
                unsigned short* __restrict__ kb_raw,
                unsigned short* __restrict__ vb_raw,
                float* __restrict__ cached_kv) {
    __shared__ float wtile[8][32 * 16];
    __bf16* qb = (__bf16*)qb_raw;
    __bf16* kb = (__bf16*)kb_raw;
    __bf16* vb = (__bf16*)vb_raw;

    const int wave = threadIdx.x >> 5;
    const int lane = threadIdx.x & 31;
    const int row_tile = blockIdx.x / 24;                 // 0..255  (M/16)
    const int col_tile = (blockIdx.x % 24) * 8 + wave;    // 0..191  (N/16)
    const int M0 = row_tile * 16;
    const int N0 = col_tile * 16;

    f32x8 acc = {};
    float* wl = wtile[wave];

    for (int k0 = 0; k0 < EMBED; k0 += 32) {
        const float* wp = Wqkv + (size_t)(k0 + lane) * QKV_N + N0;
#pragma unroll
        for (int i = 0; i < 16; i += 4)
            *(f32x4*)&wl[lane * 16 + i] = *(const f32x4*)&wp[i];
        __syncthreads();

        bf16x16 a = load_a_f32(x + (size_t)M0 * EMBED + k0, EMBED, lane);
        bf16x16 b = load_b_lds(wl, lane);
        acc = wmma_bf16(a, b, acc);
        __syncthreads();
    }

    const int qlb = (lane < 16) ? 0 : 8;
    const int n   = N0 + (lane & 15);
    const float bias = bqkv[n];
    const int sect = n >> 10;          // 0=q, 1=k, 2=v
    const int nn = n & (EMBED - 1);
    const int h = nn >> 6, d = nn & 63;
#pragma unroll
    for (int r = 0; r < 8; ++r) {
        int m = M0 + qlb + r;
        int bI = m >> 11, s = m & (SEQ - 1);
        float v = acc[r] + bias;
        size_t hs = (((size_t)bI * HEADS + h) * SEQ + s) * HDIM + d;
        if (sect == 0) {
            qb[hs] = (__bf16)v;
        } else if (sect == 1) {
            kb[hs] = (__bf16)v;
            cached_kv[((((size_t)bI * 2 + 0) * HEADS + h) * SEQ + s) * HDIM + d] = v;
        } else {
            vb[hs] = (__bf16)v;
            cached_kv[((((size_t)bI * 2 + 1) * HEADS + h) * SEQ + s) * HDIM + d] = v;
        }
    }
}

// ---------------------------------------------------------------------------
// Kernel 2: streaming scores -> head-axis softmax -> PV.
// Grid = B * (S/16) * KSPLIT blocks of 128 threads (4 waves, 4 heads each).
// Per 32-key step:
//   phase0: TDM tensor_load_to_lds of 32x64 bf16 V tiles (one per head),
//           async, overlapped with phases 1-2, tracked by TENSORcnt
//   phase1: per-head QK^T (WMMA), scale 1/8, causal mask, scores -> LDS
//   phase2: cross-head max + sum(exp) per (q,k) -> LDS
//   phase3: s_wait_tensorcnt 0; w = exp(s-max)/den as bf16 A-frag,
//           attn @ V_lds (WMMA) -> fp32 accumulators
// Epilogue: fp32 global atomicAdd into y_ws [B,H,S,D].
// ---------------------------------------------------------------------------
__global__ void __launch_bounds__(128)
attn_softmax_pv_kernel(const unsigned short* __restrict__ qb_raw,
                       const unsigned short* __restrict__ kb_raw,
                       const unsigned short* __restrict__ vb_raw,
                       float* __restrict__ y_ws) {
    __shared__ float sS[HEADS][16][32];    // scaled+masked scores [h][q][k]
    __shared__ float sMax[16][32];         // max over heads
    __shared__ float sRcp[16][32];         // 1 / sum_h exp(s - max)
#if USE_TDM
    __shared__ __align__(16) unsigned short vstage[4][4][32 * HDIM]; // 64KB
#endif

    const __bf16* qb = (const __bf16*)qb_raw;
    const __bf16* kb = (const __bf16*)kb_raw;
    const __bf16* vb = (const __bf16*)vb_raw;

    const int wave = threadIdx.x >> 5;
    const int lane = threadIdx.x & 31;
    const int bid  = blockIdx.x;
    const int kc = bid & (KSPLIT - 1);
    const int qt = (bid >> 2) & 127;
    const int b  = bid >> 9;
    const int q0 = qt * 16;

    const __bf16* qbB = qb + (size_t)b * HEADS * SEQ * HDIM;
    const __bf16* kbB = kb + (size_t)b * HEADS * SEQ * HDIM;
    const __bf16* vbB = vb + (size_t)b * HEADS * SEQ * HDIM;

    f32x8 yacc[4][4] = {};   // [head-in-wave][16-wide d tile]

    const int ksteps = (SEQ / KSPLIT) / 32;   // 16
    for (int ks = 0; ks < ksteps; ++ks) {
        const int kbase = kc * (SEQ / KSPLIT) + ks * 32;

#if USE_TDM
        // ---- phase 0: async TDM load of this wave's 4 V tiles into LDS ----
        // D# group0: count=1 | lds_addr | global_addr | type=2 (ISA 8.3)
        // D# group1: data_size=2B, tensor 64 x 32, tile 64 x 32, stride 64
        for (int hh = 0; hh < 4; ++hh) {
            const int h = wave * 4 + hh;
            const __bf16* vsrc = vbB + ((size_t)h * SEQ + kbase) * HDIM;
            unsigned lds_off =
                (unsigned)(size_t)&vstage[wave][hh][0];
            lds_off = __builtin_amdgcn_readfirstlane(lds_off);
            unsigned long long ga = (unsigned long long)(size_t)vsrc;
            unsigned ga_lo = __builtin_amdgcn_readfirstlane((unsigned)ga);
            unsigned ga_hi = __builtin_amdgcn_readfirstlane((unsigned)(ga >> 32));
            u32x4 g0 = { 1u, lds_off, ga_lo,
                         (ga_hi & 0x01FFFFFFu) | 0x80000000u };
            i32x8 g1 = { (int)(1u << 16),        // data_size = 1 -> 2 bytes
                         (int)((unsigned)HDIM << 16),  // tensor_dim0 = 64
                         (int)(32u << 16),       // tensor_dim1 = 32
                         (int)(64u << 16),       // tile_dim0 = 64
                         32,                     // tile_dim1 = 32
                         HDIM,                   // tensor_dim0_stride = 64
                         0, 0 };
            i32x4 gz = { 0, 0, 0, 0 };
#if defined(__clang_major__) && (__clang_major__ >= 23)
            i32x8 gz8 = { 0, 0, 0, 0, 0, 0, 0, 0 };
            __builtin_amdgcn_tensor_load_to_lds(g0, g1, gz, gz, gz8, 0);
#else
            __builtin_amdgcn_tensor_load_to_lds(g0, g1, gz, gz, 0);
#endif
        }
#endif

        // ---- phase 1: scores for this wave's 4 heads ----
        for (int hh = 0; hh < 4; ++hh) {
            const int h = wave * 4 + hh;
            const __bf16* qrow = qbB + ((size_t)h * SEQ + q0) * HDIM;
            const __bf16* krow = kbB + (size_t)h * SEQ * HDIM;
            if (kbase + 32 < SEQ)
                __builtin_prefetch(krow + (size_t)(kbase + 32) * HDIM, 0, 1);
#pragma unroll
            for (int sub = 0; sub < 2; ++sub) {           // 16-key subtiles
                f32x8 sacc = {};
#pragma unroll
                for (int ds = 0; ds < 2; ++ds) {          // d = 0..31, 32..63
                    bf16x16 a  = load_a_bf16(qrow + ds * 32, HDIM, lane);
                    bf16x16 bt = load_b_kT(krow, kbase + sub * 16, ds * 32, lane);
                    sacc = wmma_bf16(a, bt, sacc);
                }
                const int kl  = (lane & 15) + sub * 16;
                const int kg  = kbase + kl;
                const int qlb = (lane < 16) ? 0 : 8;
#pragma unroll
                for (int r = 0; r < 8; ++r) {
                    int qg = q0 + qlb + r;
                    float sv = (kg <= qg) ? sacc[r] * 0.125f : -10000.0f;
                    sS[h][qlb + r][kl] = sv;
                }
            }
        }
        __syncthreads();

        // ---- phase 2: cross-head max & denom per (q,k) ----
        for (int t = threadIdx.x; t < 16 * 32; t += 128) {
            const int qq = t >> 5, kk = t & 31;
            float mx = sS[0][qq][kk];
#pragma unroll
            for (int h = 1; h < HEADS; ++h) mx = fmaxf(mx, sS[h][qq][kk]);
            float den = 0.0f;
#pragma unroll
            for (int h = 0; h < HEADS; ++h) den += __expf(sS[h][qq][kk] - mx);
            sMax[qq][kk] = mx;
            sRcp[qq][kk] = 1.0f / den;
        }
        __syncthreads();

#if USE_TDM
        // All of this wave's V tiles must be resident before PV.
        __builtin_amdgcn_s_wait_tensorcnt(0);
#endif

        // ---- phase 3: attn @ V for this wave's 4 heads ----
        for (int hh = 0; hh < 4; ++hh) {
            const int h = wave * 4 + hh;
            bf16x16 aw;
            {
                const int m   = lane & 15;
                const int klo = (lane < 16) ? 0 : 16;
#pragma unroll
                for (int i = 0; i < 8; ++i) {
                    int k1 = klo + i, k2 = klo + 8 + i;
                    aw[i]     = (__bf16)(__expf(sS[h][m][k1] - sMax[m][k1]) * sRcp[m][k1]);
                    aw[8 + i] = (__bf16)(__expf(sS[h][m][k2] - sMax[m][k2]) * sRcp[m][k2]);
                }
            }
#if USE_TDM
            const __bf16* vtile = (const __bf16*)&vstage[wave][hh][0];
#else
            const __bf16* vtile = vbB + ((size_t)h * SEQ + kbase) * HDIM;
#endif
#pragma unroll
            for (int dt = 0; dt < 4; ++dt) {
                bf16x16 bv = load_b_v_tile(vtile, dt * 16, lane);
                yacc[hh][dt] = wmma_bf16(aw, bv, yacc[hh][dt]);
            }
        }
        __syncthreads();
    }

    // ---- epilogue: accumulate partial y into y_ws [B,H,S,D] ----
    const int qlb = (lane < 16) ? 0 : 8;
    const int dl  = lane & 15;
    for (int hh = 0; hh < 4; ++hh) {
        const int h = wave * 4 + hh;
#pragma unroll
        for (int dt = 0; dt < 4; ++dt) {
            const int d = dt * 16 + dl;
#pragma unroll
            for (int r = 0; r < 8; ++r) {
                int qg = q0 + qlb + r;
                atomicAdd(&y_ws[(((size_t)b * HEADS + h) * SEQ + qg) * HDIM + d],
                          yacc[hh][dt][r]);
            }
        }
    }
}

// ---------------------------------------------------------------------------
// Kernel 3: out_y = y @ W_dense + b_dense.
// y lives in y_ws with layout [B,H,S,D]; logical K index e = h*64+d, so each
// 32-wide K step stays inside one head chunk (contiguous rows, stride 64).
// ---------------------------------------------------------------------------
__global__ void __launch_bounds__(256)
dense_gemm_kernel(const float* __restrict__ y_ws, const float* __restrict__ Wd,
                  const float* __restrict__ bd, float* __restrict__ out) {
    __shared__ float wtile[8][32 * 16];
    const int wave = threadIdx.x >> 5;
    const int lane = threadIdx.x & 31;
    const int row_tile = blockIdx.x / 8;                 // 0..255
    const int col_tile = (blockIdx.x % 8) * 8 + wave;    // 0..63
    const int M0 = row_tile * 16;
    const int N0 = col_tile * 16;

    f32x8 acc = {};
    float* wl = wtile[wave];

    for (int k0 = 0; k0 < EMBED; k0 += 32) {
        const float* wp = Wd + (size_t)(k0 + lane) * EMBED + N0;
#pragma unroll
        for (int i = 0; i < 16; i += 4)
            *(f32x4*)&wl[lane * 16 + i] = *(const f32x4*)&wp[i];
        __syncthreads();

        const int h = k0 >> 6, off = k0 & 63;
        const int bI = M0 >> 11, s0 = M0 & (SEQ - 1);
        const float* abase =
            y_ws + (((size_t)bI * HEADS + h) * SEQ + s0) * HDIM + off;
        bf16x16 a = load_a_f32(abase, HDIM, lane);
        bf16x16 b = load_b_lds(wl, lane);
        acc = wmma_bf16(a, b, acc);
        __syncthreads();
    }

    const int qlb = (lane < 16) ? 0 : 8;
    const int n   = N0 + (lane & 15);
    const float bias = bd[n];
#pragma unroll
    for (int r = 0; r < 8; ++r) {
        int m = M0 + qlb + r;
        out[(size_t)m * EMBED + n] = acc[r] + bias;
    }
}

// ---------------------------------------------------------------------------
// Launcher. Workspace layout (bytes):
//   [0      , 8MB )  q  bf16 [B,H,S,D]
//   [8MB    , 16MB)  k  bf16 [B,H,S,D]
//   [16MB   , 24MB)  v  bf16 [B,H,S,D]
//   [24MB   , 40MB)  y  fp32 [B,H,S,D] (pre-dense attention output)
// d_out: y [B,S,E] (4,194,304 f32) then cached_kv [B,2,H,S,D] (8,388,608 f32)
// ---------------------------------------------------------------------------
extern "C" void kernel_launch(void* const* d_in, const int* in_sizes, int n_in,
                              void* d_out, int out_size, void* d_ws, size_t ws_size,
                              hipStream_t stream) {
    const float* x    = (const float*)d_in[0];
    const float* Wqkv = (const float*)d_in[1];
    const float* bqkv = (const float*)d_in[2];
    const float* Wd   = (const float*)d_in[3];
    const float* bd   = (const float*)d_in[4];

    float* out_y   = (float*)d_out;
    float* cached  = out_y + (size_t)BATCH * SEQ * EMBED;

    char* ws = (char*)d_ws;
    unsigned short* qb = (unsigned short*)(ws);
    unsigned short* kb = (unsigned short*)(ws + (8ull << 20));
    unsigned short* vb = (unsigned short*)(ws + (16ull << 20));
    float*        y_ws = (float*)(ws + (24ull << 20));

    hipMemsetAsync(y_ws, 0, (size_t)BATCH * HEADS * SEQ * HDIM * sizeof(float),
                   stream);

    qkv_gemm_kernel<<<dim3(256 * 24), dim3(256), 0, stream>>>(
        x, Wqkv, bqkv, qb, kb, vb, cached);

    attn_softmax_pv_kernel<<<dim3(BATCH * (SEQ / 16) * KSPLIT), dim3(128), 0,
                             stream>>>(qb, kb, vb, y_ws);

    dense_gemm_kernel<<<dim3(256 * 8), dim3(256), 0, stream>>>(
        y_ws, Wd, bd, out_y);
}
